// SliddingWindowAttention_37976100831413
// MI455X (gfx1250) — matrix operand order repaired and checked by reading, not compiled, not run on hardware
//
#include <hip/hip_runtime.h>

#define BB 2
#define NN 4096
#define EE 512
#define DD 64
#define MTOT (BB*NN)
#define WSZ 129
#define TAILN (NN - WSZ)
#define AW 2   // waves per attention block

typedef unsigned short u16;
typedef unsigned int   u32;
typedef __bf16 bf16;
typedef bf16  v16bf __attribute__((ext_vector_type(16)));
typedef float v8f   __attribute__((ext_vector_type(8)));
typedef u32   u32x4 __attribute__((ext_vector_type(4)));

union Frag { v16bf v; u32x4 q[2]; };

__device__ __forceinline__ u16 f2bf(float f) {
  u32 u = __float_as_uint(f);
  u += 0x7FFFu + ((u >> 16) & 1u);   // round-to-nearest-even
  return (u16)(u >> 16);
}

// packed f32x2 -> bf16x2 (single v_cvt_pk_bf16_f32 when available)
__device__ __forceinline__ u32 pack2(float lo, float hi) {
#if __has_builtin(__builtin_amdgcn_cvt_pk_bf16_f32)
  auto r = __builtin_amdgcn_cvt_pk_bf16_f32(lo, hi);
  u32 out;
  __builtin_memcpy(&out, &r, sizeof(out));
  return out;
#else
  return (u32)f2bf(lo) | ((u32)f2bf(hi) << 16);
#endif
}

__device__ __forceinline__ v8f wmma_bf16(const Frag& a, const Frag& b, v8f c) {
  return __builtin_amdgcn_wmma_f32_16x16x32_bf16(false, a.v, false, b.v,
                                                 (short)0, c, false, false);
}

// CDNA5 async global->LDS copy (16B per lane), tracked by ASYNCcnt
__device__ __forceinline__ void async_copy16(u32 lds_off, const u16* gsrc) {
  asm volatile("global_load_async_to_lds_b128 %0, %1, off"
               :: "v"(lds_off), "v"(gsrc) : "memory");
}
__device__ __forceinline__ void wait_async0() {
#if __has_builtin(__builtin_amdgcn_s_wait_asynccnt)
  __builtin_amdgcn_s_wait_asynccnt(0);
#else
  asm volatile("s_wait_asynccnt 0x0" ::: "memory");
#endif
}

// ---------------- weight transpose to bf16 ----------------
// WT[mat][n][c] = W_mat[c][n]
__global__ void conv_w_kernel(const float* __restrict__ Wq, const float* __restrict__ Wk,
                              const float* __restrict__ Wv, u16* __restrict__ WT) {
  int idx = blockIdx.x * blockDim.x + threadIdx.x;
  if (idx >= 3 * DD * EE) return;
  int mat = idx / (DD * EE);
  int rem = idx - mat * (DD * EE);
  int n = rem / EE;
  int c = rem - n * EE;
  const float* W = (mat == 0) ? Wq : (mat == 1) ? Wk : Wv;
  WT[idx] = f2bf(W[(size_t)c * DD + n]);
}

// ---------------- projection GEMM (Q,K,V), fused x->bf16 ----------------
// 4 waves/block, 64 rows/block; weights double-buffered in LDS via async copy.
__global__ void __launch_bounds__(128) proj_kernel(
    const float* __restrict__ x, const u16* __restrict__ WT,
    const float* __restrict__ bq, const float* __restrict__ bk, const float* __restrict__ bv,
    u16* __restrict__ Qbf, u16* __restrict__ Kbf, float* __restrict__ Vf) {
  __shared__ __align__(16) u16 Bs[2][3][DD][40];   // rows padded 32->40 halves (bank-spread)

  const int tid  = threadIdx.x;
  const int w    = tid >> 5;
  const int lane = tid & 31;
  const int m  = lane & 15;
  const int hi = lane >> 4;
  const int row0 = blockIdx.x * 64 + w * 16;

  v8f acc[3][4];
  #pragma unroll
  for (int t = 0; t < 3; ++t)
    #pragma unroll
    for (int nt = 0; nt < 4; ++nt) acc[t][nt] = {};

  // stage one 32-wide K slice of all three weight matrices: 768 16B chunks
  auto stage = [&](int buf, int kk) {
    #pragma unroll
    for (int it = 0; it < 6; ++it) {
      int c   = tid + 128 * it;
      int mat = c >> 8;          // 256 chunks per matrix
      int rem = c & 255;
      int n   = rem >> 2;
      int cc  = rem & 3;
      async_copy16((u32)(uintptr_t)&Bs[buf][mat][n][cc * 8],
                   WT + ((size_t)mat * DD + n) * EE + kk + cc * 8);
    }
  };

  const float* arow = x + (size_t)(row0 + m) * EE;
  stage(0, 0);

  for (int ks = 0; ks < 16; ++ks) {
    const int kk  = ks * 32;
    const int buf = ks & 1;
    wait_async0();
    __syncthreads();                       // staged slice visible to all waves
    if (ks + 1 < 16) stage(buf ^ 1, kk + 32);
    if (kk + 128 < EE) __builtin_prefetch((const void*)(arow + kk + 128), 0, 0);

    // A fragment: f32 x -> bf16, ISA A-layout (runs [0..7] and [16..23] per lane-half)
    Frag a;
    {
      const float4* pa = (const float4*)(arow + kk + hi * 8);
      float4 f0 = pa[0], f1 = pa[1], f2 = pa[4], f3 = pa[5];
      u32x4 q0 = { pack2(f0.x, f0.y), pack2(f0.z, f0.w),
                   pack2(f1.x, f1.y), pack2(f1.z, f1.w) };
      u32x4 q1 = { pack2(f2.x, f2.y), pack2(f2.z, f2.w),
                   pack2(f3.x, f3.y), pack2(f3.z, f3.w) };
      a.q[0] = q0; a.q[1] = q1;
    }

    #pragma unroll
    for (int mat = 0; mat < 3; ++mat) {
      #pragma unroll
      for (int nt = 0; nt < 4; ++nt) {
        Frag bf;
        const u32x4* pb = (const u32x4*)(&Bs[buf][mat][nt * 16 + m][0] + hi * 8);
        bf.q[0] = pb[0]; bf.q[1] = pb[2];
        acc[mat][nt] = wmma_bf16(a, bf, acc[mat][nt]);
      }
    }
    __syncthreads();                       // all waves done with buf before re-stage
  }

  #pragma unroll
  for (int nt = 0; nt < 4; ++nt) {
    int d = nt * 16 + m;
    float biasq = bq[d], biask = bk[d], biasv = bv[d];
    #pragma unroll
    for (int v = 0; v < 8; ++v) {
      int row = row0 + v + 8 * hi;         // C/D layout: M = v + 8*(lane>=16)
      Qbf[(size_t)row * DD + d] = f2bf(acc[0][nt][v] + biasq);
      Kbf[(size_t)row * DD + d] = f2bf(acc[1][nt][v] + biask);
      Vf [(size_t)row * DD + d] = acc[2][nt][v] + biasv;
    }
  }
}

// ---------------- Vtail = sum of V rows 129..4095 per batch ----------------
__global__ void vtail_kernel(const float* __restrict__ Vf, float* __restrict__ Vtail) {
  __shared__ float part[4][DD];
  int b = blockIdx.x;
  int d = threadIdx.x & 63;
  int c = threadIdx.x >> 6;
  float s = 0.0f;
  for (int o = WSZ + c; o < NN; o += 4) s += Vf[((size_t)b * NN + o) * DD + d];
  part[c][d] = s;
  __syncthreads();
  if (threadIdx.x < DD)
    Vtail[b * DD + threadIdx.x] =
        part[0][threadIdx.x] + part[1][threadIdx.x] + part[2][threadIdx.x] + part[3][threadIdx.x];
}

// ---------------- banded scores + softmax + PV ----------------
__global__ void __launch_bounds__(32 * AW) attn_kernel(
    const u16* __restrict__ Qbf, const u16* __restrict__ Kbf,
    const float* __restrict__ Vf, const float* __restrict__ Vtail,
    float* __restrict__ out) {
  __shared__ __align__(16) u16   VT[DD][160];        // V^T (bf16), rows o=0..128 (+zero pad)
  __shared__ __align__(16) float S[AW][16][160];     // per-wave 16x144 scores
  __shared__ __align__(16) u16   WL[AW][16][160];    // per-wave softmax weights (bf16)
  __shared__ float rENeg[AW][16];
  __shared__ float rInvZ[AW][16];

  const int b    = blockIdx.y;
  const int w    = threadIdx.x >> 5;
  const int lane = threadIdx.x & 31;
  const int m    = lane & 15;
  const int hi   = lane >> 4;
  const int i0   = (blockIdx.x * AW + w) * 16;

  for (int idx = threadIdx.x; idx < DD * 160; idx += blockDim.x) {
    int o = idx / DD;
    int d = idx - o * DD;
    VT[d][o] = (o < WSZ) ? f2bf(Vf[((size_t)b * NN + o) * DD + d]) : (u16)0;
  }
  __syncthreads();

  Frag A0, A1;
  const u16* qrow = Qbf + ((size_t)(b * NN + i0 + m)) * DD;
  {
    const u32x4* p0 = (const u32x4*)(qrow + hi * 8);
    const u32x4* p1 = (const u32x4*)(qrow + 32 + hi * 8);
    A0.q[0] = p0[0]; A0.q[1] = p0[2];
    A1.q[0] = p1[0]; A1.q[1] = p1[2];
  }

  const int j0base = i0 - 64;

  #pragma unroll
  for (int jt = 0; jt < 9; ++jt) {
    int j = j0base + jt * 16 + m;
    int jc = j < 0 ? 0 : (j > NN - 1 ? NN - 1 : j);   // clamp; masked later
    const u16* krow = Kbf + ((size_t)(b * NN + jc)) * DD;
    Frag B0, B1;
    const u32x4* p0 = (const u32x4*)(krow + hi * 8);
    const u32x4* p1 = (const u32x4*)(krow + 32 + hi * 8);
    B0.q[0] = p0[0]; B0.q[1] = p0[2];
    B1.q[0] = p1[0]; B1.q[1] = p1[2];
    v8f acc = {};
    acc = wmma_bf16(A0, B0, acc);
    acc = wmma_bf16(A1, B1, acc);
    #pragma unroll
    for (int v = 0; v < 8; ++v) S[w][v + 8 * hi][jt * 16 + m] = acc[v];
  }

  for (int r = 0; r < 16; ++r) {
    float sv[5];
    float mx = 0.0f;                                   // implicit zero columns
    #pragma unroll
    for (int t = 0; t < 5; ++t) {
      int o = lane + 32 * t;
      bool act = (o < WSZ);
      int jl = o + r; if (jl > 159) jl = 159;
      int j = j0base + o + r;
      bool val = act && (j >= 0) && (j < NN);
      float s = val ? S[w][r][jl] : 0.0f;              // invalid j -> score 0 (ref semantics)
      sv[t] = act ? s : -3.0e38f;
      mx = fmaxf(mx, sv[t]);
    }
    #pragma unroll
    for (int off = 16; off > 0; off >>= 1) mx = fmaxf(mx, __shfl_xor(mx, off, 32));

    float sum = 0.0f, ev[5];
    #pragma unroll
    for (int t = 0; t < 5; ++t) {
      int o = lane + 32 * t;
      float e = (o < WSZ) ? __expf(sv[t] - mx) : 0.0f;
      ev[t] = e; sum += e;
    }
    #pragma unroll
    for (int off = 16; off > 0; off >>= 1) sum += __shfl_xor(sum, off, 32);

    float eneg = __expf(-mx);
    #pragma unroll
    for (int t = 0; t < 5; ++t) {
      int o = lane + 32 * t;
      if (o < WSZ) WL[w][r][o] = f2bf(ev[t]);
    }
    if (lane < 160 - WSZ) WL[w][r][WSZ + lane] = 0;
    if (lane == 0) {
      rENeg[w][r] = eneg;
      rInvZ[w][r] = 1.0f / (sum + (float)TAILN * eneg);
    }
  }

  v8f acc[4];
  #pragma unroll
  for (int nt = 0; nt < 4; ++nt) acc[nt] = {};
  #pragma unroll
  for (int kk = 0; kk < 160; kk += 32) {
    Frag a;
    const u32x4* pa = (const u32x4*)(&WL[w][m][0] + kk + hi * 8);
    a.q[0] = pa[0]; a.q[1] = pa[2];
    #pragma unroll
    for (int nt = 0; nt < 4; ++nt) {
      Frag bf;
      const u32x4* pb = (const u32x4*)(&VT[nt * 16 + m][0] + kk + hi * 8);
      bf.q[0] = pb[0]; bf.q[1] = pb[2];
      acc[nt] = wmma_bf16(a, bf, acc[nt]);
    }
  }

  #pragma unroll
  for (int nt = 0; nt < 4; ++nt) {
    int d = nt * 16 + m;
    float vt = Vtail[b * DD + d];
    #pragma unroll
    for (int v = 0; v < 8; ++v) {
      int M = v + 8 * hi;
      out[((size_t)(b * NN + i0 + M)) * DD + d] =
          (acc[nt][v] + rENeg[w][M] * vt) * rInvZ[w][M];
    }
  }
}

// ---------------- launcher ----------------
extern "C" void kernel_launch(void* const* d_in, const int* in_sizes, int n_in,
                              void* d_out, int out_size, void* d_ws, size_t ws_size,
                              hipStream_t stream) {
  (void)in_sizes; (void)n_in; (void)out_size; (void)ws_size;
  const float* x  = (const float*)d_in[0];
  const float* Wq = (const float*)d_in[1];
  const float* bq = (const float*)d_in[2];
  const float* Wk = (const float*)d_in[3];
  const float* bk = (const float*)d_in[4];
  const float* Wv = (const float*)d_in[5];
  const float* bv = (const float*)d_in[6];
  float* out = (float*)d_out;

  // workspace layout (~4.4 MB)
  u16* WT  = (u16*)d_ws;                            // 3*64*512 bf16 (transposed)
  u16* Qbf = WT  + (size_t)3 * DD * EE;             // 8192*64 bf16
  u16* Kbf = Qbf + (size_t)MTOT * DD;               // 8192*64 bf16
  float* Vf    = (float*)(Kbf + (size_t)MTOT * DD); // 8192*64 f32
  float* Vtail = Vf + (size_t)MTOT * DD;            // 2*64 f32

  conv_w_kernel<<<(3 * DD * EE + 255) / 256, 256, 0, stream>>>(Wq, Wk, Wv, WT);
  proj_kernel<<<MTOT / 64, 128, 0, stream>>>(x, WT, bq, bk, bv, Qbf, Kbf, Vf);
  vtail_kernel<<<BB, 256, 0, stream>>>(Vf, Vtail);
  attn_kernel<<<dim3(NN / (16 * AW), BB), 32 * AW, 0, stream>>>(Qbf, Kbf, Vf, Vtail, out);
}
